// MoE_33552284517106
// MI455X (gfx1250) — compile-verified, matrix operand order checked by reading, not run on hardware
//
#include <hip/hip_runtime.h>
#include <hip/hip_bf16.h>

// ---------------- constants ----------------
#define CCH    256          // channels
#define DWCH   512          // expanded channels
#define HW     4096         // 64*64
#define WIDTH  64
#define NB     32           // batch
#define ASTR   264          // padded A row stride (halves)
#define BSTR   264          // padded B row stride (halves)
#define OSTR   65           // padded out staging row stride (floats)

typedef __bf16 bf16;
typedef __bf16 v16bf __attribute__((ext_vector_type(16)));
typedef __bf16 v8bf  __attribute__((ext_vector_type(8)));
typedef float  v8f   __attribute__((ext_vector_type(8)));

// LDS byte layout for the GEMM kernels
#define A_OFF   0                         // 64*264*2  = 33792
#define B_OFF   33792                     // 128*264*2 = 67584
#define O_OFF   101376                    // 128*65*4  = 33280
#define S_OFF   134656                    // scalars
#define LNG_SMEM  139776
#define TAIL_SMEM 136704

// ---------------- helpers ----------------
__device__ __forceinline__ v16bf load_frag_lds(const bf16* row, int kb) {
  // A/B 16-bit fragment layout (ISA 7.12.2): lane<16 holds K = kb..kb+7 and
  // kb+16..kb+23 (kb = ktile base for lanes<16, ktile+8 for lanes>=16).
  v8bf lo = *(const v8bf*)(row + kb);
  v8bf hi = *(const v8bf*)(row + kb + 16);
  v16bf r;
#pragma unroll
  for (int i = 0; i < 8; ++i) { r[i] = lo[i]; r[i + 8] = hi[i]; }
  return r;
}

__device__ __forceinline__ unsigned pack2bf(float a, float b) {
  union { bf16 h[2]; unsigned u; } p;
  p.h[0] = (bf16)a; p.h[1] = (bf16)b;
  return p.u;
}

// ---------------- utility kernels ----------------
__global__ __launch_bounds__(256) void k_zero(float* __restrict__ p, size_t n) {
  size_t i = (size_t)blockIdx.x * blockDim.x + threadIdx.x;
  size_t st = (size_t)gridDim.x * blockDim.x;
  for (; i < n; i += st) p[i] = 0.0f;
}

// gx[b*256+c] = mean over HW of x[b][c][:]
__global__ __launch_bounds__(256) void k_gx(const float* __restrict__ x,
                                            float* __restrict__ gx) {
  __shared__ float red[256];
  const int tid = threadIdx.x;
  const float* p = x + (size_t)blockIdx.x * HW;
  float s = 0.0f;
#pragma unroll
  for (int j = 0; j < 16; ++j) s += p[tid + j * 256];
  red[tid] = s;
  __syncthreads();
  for (int st = 128; st > 0; st >>= 1) {
    if (tid < st) red[tid] += red[tid + st];
    __syncthreads();
  }
  if (tid == 0) gx[blockIdx.x] = red[0] * (1.0f / (float)HW);
}

// gates[b*3+e]: top-2 softmax of gx[b] @ w_gate  (w_gate is [C][3] row-major)
__global__ __launch_bounds__(256) void k_gates(const float* __restrict__ gx,
                                               const float* __restrict__ wg,
                                               float* __restrict__ gates) {
  __shared__ float r0[256], r1[256], r2[256];
  const int tid = threadIdx.x, b = blockIdx.x;
  float v = gx[b * 256 + tid];
  r0[tid] = v * wg[tid * 3 + 0];
  r1[tid] = v * wg[tid * 3 + 1];
  r2[tid] = v * wg[tid * 3 + 2];
  __syncthreads();
  for (int st = 128; st > 0; st >>= 1) {
    if (tid < st) {
      r0[tid] += r0[tid + st];
      r1[tid] += r1[tid + st];
      r2[tid] += r2[tid + st];
    }
    __syncthreads();
  }
  if (tid == 0) {
    float l[3] = { r0[0], r1[0], r2[0] };
    int i0 = 0;
    if (l[1] > l[i0]) i0 = 1;
    if (l[2] > l[i0]) i0 = 2;
    int i1 = -1;
    for (int j = 0; j < 3; ++j) {
      if (j == i0) continue;
      if (i1 < 0 || l[j] > l[i1]) i1 = j;
    }
    float e1 = __expf(l[i1] - l[i0]);
    float g0 = 1.0f / (1.0f + e1);
    float g1 = e1 * g0;
    float o[3] = { 0.0f, 0.0f, 0.0f };
    o[i0] = g0; o[i1] = g1;
    gates[b * 3 + 0] = o[0];
    gates[b * 3 + 1] = o[1];
    gates[b * 3 + 2] = o[2];
  }
}

// ---------------- LN + conv1x1 (GEMM 256 -> O) -> bf16 ----------------
// grid = (131072/64, O/128); block = 256 (8 waves)
__global__ __launch_bounds__(256) void k_ln_gemm(
    const float* __restrict__ xin,   // [B,256,HW] f32
    const float* __restrict__ lnw, const float* __restrict__ lnb,
    const float* __restrict__ wgt,   // [O,256] f32 row-major
    const float* __restrict__ bias,  // [O]
    bf16* __restrict__ tout,         // [B,O,HW] bf16
    int O) {
  extern __shared__ char smem[];
  bf16*  a_s   = (bf16*)(smem + A_OFF);
  bf16*  b_s   = (bf16*)(smem + B_OFF);
  float* out_s = (float*)(smem + O_OFF);
  float* ssum  = (float*)(smem + S_OFF);       // 256
  float* ssq   = ssum + 256;                   // 256
  float* smu   = ssq + 256;                    // 64
  float* srs   = smu + 64;                     // 64
  float* slnw  = srs + 64;                     // 256
  float* slnb  = slnw + 256;                   // 256
  float* sbias = slnb + 256;                   // 128

  const int tid = threadIdx.x;
  const int m0 = blockIdx.x * 64;      // global pixel
  const int b  = m0 >> 12;
  const int p0 = m0 & (HW - 1);
  const int n0 = blockIdx.y * 128;

  // prefetch weight tile while we do LN stats (global_prefetch_b8)
  __builtin_prefetch(wgt + (size_t)(n0 + (tid & 127)) * CCH, 0, 0);

  slnw[tid] = lnw[tid];
  slnb[tid] = lnb[tid];
  if (tid < 128) sbias[tid] = bias[n0 + tid];

  // phase 1: LN stats over 256 channels for 64 pixels
  const int i = tid & 63, q = tid >> 6;
  const float* xp = xin + (size_t)b * CCH * HW + p0 + i;
  float s1 = 0.0f, s2 = 0.0f;
#pragma unroll 4
  for (int j = 0; j < 64; ++j) {
    float v = xp[(size_t)(q * 64 + j) * HW];
    s1 += v; s2 += v * v;
  }
  ssum[q * 64 + i] = s1;
  ssq [q * 64 + i] = s2;
  __syncthreads();
  if (tid < 64) {
    float su = ssum[tid] + ssum[64 + tid] + ssum[128 + tid] + ssum[192 + tid];
    float sq = ssq [tid] + ssq [64 + tid] + ssq [128 + tid] + ssq [192 + tid];
    float mu = su * (1.0f / 256.0f);
    float var = sq * (1.0f / 256.0f) - mu * mu;
    smu[tid] = mu;
    srs[tid] = rsqrtf(var + 1e-6f);
  }
  __syncthreads();

  // phase 2: build A tile (LN applied, bf16). x re-read hits L2 (192MB).
  {
    float mu = smu[i], rs = srs[i];
    for (int j = 0; j < 64; ++j) {
      int c = q * 64 + j;
      float v = xp[(size_t)c * HW];
      a_s[i * ASTR + c] = (bf16)((v - mu) * rs * slnw[c] + slnb[c]);
    }
  }
  // phase 3: B tile (128 rows of W, 256 K each), f32 -> bf16
  for (int k = 0; k < 128; ++k) {
    int e = tid + k * 256;
    int row = e >> 8, col = e & 255;
    b_s[row * BSTR + col] = (bf16)wgt[(size_t)(n0 + row) * CCH + col];
  }
  __syncthreads();

  // phase 4: WMMA. wave w -> Msub = w&3 (16 pixels), Nsubs = (w>>2)*4 .. +3
  const int wv = tid >> 5, ln = tid & 31;
  const int msub = wv & 3, nh = wv >> 2;
  const int lm = ln & 15;
  const int khalf = (ln < 16) ? 0 : 8;
  v8f acc[4];
#pragma unroll
  for (int j = 0; j < 4; ++j)
#pragma unroll
    for (int e = 0; e < 8; ++e) acc[j][e] = 0.0f;

  const bf16* arow = a_s + (size_t)(msub * 16 + lm) * ASTR;
#pragma unroll
  for (int kk = 0; kk < 256; kk += 32) {
    v16bf af = load_frag_lds(arow, kk + khalf);
#pragma unroll
    for (int j = 0; j < 4; ++j) {
      const bf16* brow = b_s + (size_t)((nh * 4 + j) * 16 + lm) * BSTR;
      v16bf bfv = load_frag_lds(brow, kk + khalf);
      acc[j] = __builtin_amdgcn_wmma_f32_16x16x32_bf16(
          false, af, false, bfv, (short)0, acc[j], false, false);
    }
  }

  // phase 5: stage D (f32) -> coalesced bf16 store
#pragma unroll
  for (int j = 0; j < 4; ++j) {
    int oloc = (nh * 4 + j) * 16 + lm;
#pragma unroll
    for (int r = 0; r < 8; ++r) {
      int ml = msub * 16 + r + ((ln < 16) ? 0 : 8);
      out_s[oloc * OSTR + ml] = acc[j][r];
    }
  }
  __syncthreads();
  {
    int oloc = tid >> 1;
    int mst = (tid & 1) * 32;
    bf16* dst = tout + (size_t)b * O * HW + (size_t)(n0 + oloc) * HW + p0 + mst;
    float bia = sbias[oloc];
#pragma unroll
    for (int m = 0; m < 32; m += 2) {
      float v0 = out_s[oloc * OSTR + mst + m] + bia;
      float v1 = out_s[oloc * OSTR + mst + m + 1] + bia;
      *(unsigned*)(dst + m) = pack2bf(v0, v1);
    }
  }
}

// ---------------- dwconv3x3 + SimpleGate + pool partial ----------------
// grid = (B*256, 16); block = 256; one (b,c) channel per blockIdx.x
__global__ __launch_bounds__(256) void k_dwgate(
    const bf16* __restrict__ t,     // [B,512,HW] bf16
    const float* __restrict__ w2,   // [512,1,3,3]
    const float* __restrict__ b2,   // [512]
    bf16* __restrict__ hbuf,        // [B,256,HW] bf16
    float* __restrict__ pool) {     // [B*256] (pre-zeroed) sum
  __shared__ float red[256];
  const int tid = threadIdx.x;
  const int bc = blockIdx.x;
  const int b = bc >> 8, c = bc & 255;
  const int p = blockIdx.y * 256 + tid;
  const int hh = p >> 6, ww = p & (WIDTH - 1);

  const bf16* t1 = t + ((size_t)b * DWCH + c) * HW;
  const bf16* t2 = t + ((size_t)b * DWCH + c + 256) * HW;
  const float* k1 = w2 + (size_t)c * 9;
  const float* k2 = w2 + (size_t)(c + 256) * 9;

  float s1 = b2[c], s2 = b2[c + 256];
#pragma unroll
  for (int ky = 0; ky < 3; ++ky) {
    int ih = hh + ky - 1;
    if (ih < 0 || ih >= WIDTH) continue;
#pragma unroll
    for (int kx = 0; kx < 3; ++kx) {
      int iw = ww + kx - 1;
      if (iw < 0 || iw >= WIDTH) continue;
      int ip = ih * WIDTH + iw;
      s1 += k1[ky * 3 + kx] * (float)t1[ip];
      s2 += k2[ky * 3 + kx] * (float)t2[ip];
    }
  }
  float val = s1 * s2;
  hbuf[((size_t)b * CCH + c) * HW + p] = (bf16)val;

  red[tid] = val;
  __syncthreads();
  for (int st = 128; st > 0; st >>= 1) {
    if (tid < st) red[tid] += red[tid + st];
    __syncthreads();
  }
  if (tid == 0) atomicAdd(&pool[bc], red[0]);
}

// ---------------- SCA: att[b][o] = bsca[o] + wsca @ (pool/HW) ----------------
__global__ __launch_bounds__(256) void k_sca(const float* __restrict__ pool,
                                             const float* __restrict__ wsca,
                                             const float* __restrict__ bsca,
                                             float* __restrict__ att) {
  __shared__ float sp[256];
  const int tid = threadIdx.x, b = blockIdx.x;
  sp[tid] = pool[b * 256 + tid] * (1.0f / (float)HW);
  __syncthreads();
  float acc = bsca[tid];
  const float* wr = wsca + (size_t)tid * 256;
#pragma unroll 8
  for (int c = 0; c < 256; ++c) acc += wr[c] * sp[c];
  att[b * 256 + tid] = acc;
}

// ---------------- tail GEMM (256 -> 256) with fused epilogues ----------------
// amode: 0 = A from h_buf scaled by att[b][c];  1 = A = t[c]*t[c+256] (SimpleGate)
// omode: 0 = out += g*(D+b);  1 = y = aux + svec[o]*(D+b);
//        2 = out += g*(aux + svec[o]*(D+b))
__global__ __launch_bounds__(256) void k_tail_gemm(
    const bf16* __restrict__ ain,
    const float* __restrict__ att,    // [B*256] or null
    const float* __restrict__ wgt,    // [256,256]
    const float* __restrict__ bias,   // [256]
    const float* __restrict__ gates,  // [B*3]
    const float* __restrict__ aux,    // x or y (f32) for omode 1/2
    const float* __restrict__ svec,   // beta/gamma [256] or null
    float* __restrict__ outp,
    int amode, int omode, int eidx) {
  extern __shared__ char smem[];
  bf16*  a_s   = (bf16*)(smem + A_OFF);
  bf16*  b_s   = (bf16*)(smem + B_OFF);
  float* out_s = (float*)(smem + O_OFF);
  float* satt  = (float*)(smem + S_OFF);   // 256
  float* sbias = satt + 256;               // 128
  float* ssca  = sbias + 128;              // 128

  const int tid = threadIdx.x;
  const int m0 = blockIdx.x * 64;
  const int b  = m0 >> 12;
  const int p0 = m0 & (HW - 1);
  const int n0 = blockIdx.y * 128;

  __builtin_prefetch(wgt + (size_t)(n0 + (tid & 127)) * CCH, 0, 0);

  if (amode == 0) satt[tid] = att[b * 256 + tid];
  if (tid < 128) {
    sbias[tid] = bias[n0 + tid];
    ssca[tid] = (svec != nullptr) ? svec[n0 + tid] : 0.0f;
  }
  __syncthreads();

  // A tile
  const int i = tid & 63, q = tid >> 6;
  if (amode == 0) {
    const bf16* base = ain + (size_t)b * CCH * HW + p0 + i;
    for (int j = 0; j < 64; ++j) {
      int c = q * 64 + j;
      a_s[i * ASTR + c] = (bf16)((float)base[(size_t)c * HW] * satt[c]);
    }
  } else {
    const bf16* base = ain + (size_t)b * DWCH * HW + p0 + i;
    for (int j = 0; j < 64; ++j) {
      int c = q * 64 + j;
      float v = (float)base[(size_t)c * HW] * (float)base[(size_t)(c + 256) * HW];
      a_s[i * ASTR + c] = (bf16)v;
    }
  }
  // B tile
  for (int k = 0; k < 128; ++k) {
    int e = tid + k * 256;
    int row = e >> 8, col = e & 255;
    b_s[row * BSTR + col] = (bf16)wgt[(size_t)(n0 + row) * CCH + col];
  }
  __syncthreads();

  const int wv = tid >> 5, ln = tid & 31;
  const int msub = wv & 3, nh = wv >> 2;
  const int lm = ln & 15;
  const int khalf = (ln < 16) ? 0 : 8;
  v8f acc[4];
#pragma unroll
  for (int j = 0; j < 4; ++j)
#pragma unroll
    for (int e = 0; e < 8; ++e) acc[j][e] = 0.0f;

  const bf16* arow = a_s + (size_t)(msub * 16 + lm) * ASTR;
#pragma unroll
  for (int kk = 0; kk < 256; kk += 32) {
    v16bf af = load_frag_lds(arow, kk + khalf);
#pragma unroll
    for (int j = 0; j < 4; ++j) {
      const bf16* brow = b_s + (size_t)((nh * 4 + j) * 16 + lm) * BSTR;
      v16bf bfv = load_frag_lds(brow, kk + khalf);
      acc[j] = __builtin_amdgcn_wmma_f32_16x16x32_bf16(
          false, af, false, bfv, (short)0, acc[j], false, false);
    }
  }

#pragma unroll
  for (int j = 0; j < 4; ++j) {
    int oloc = (nh * 4 + j) * 16 + lm;
#pragma unroll
    for (int r = 0; r < 8; ++r) {
      int ml = msub * 16 + r + ((ln < 16) ? 0 : 8);
      out_s[oloc * OSTR + ml] = acc[j][r];
    }
  }
  __syncthreads();
  {
    int oloc = tid >> 1;
    int mst = (tid & 1) * 32;
    size_t g = (size_t)b * CCH * HW + (size_t)(n0 + oloc) * HW + p0 + mst;
    float bia = sbias[oloc];
    float sv = ssca[oloc];
    float sg = gates[b * 3 + eidx];
#pragma unroll
    for (int m = 0; m < 32; ++m) {
      float dv = out_s[oloc * OSTR + mst + m] + bia;
      if (omode == 0) {
        outp[g + m] += sg * dv;
      } else if (omode == 1) {
        outp[g + m] = aux[g + m] + sv * dv;
      } else {
        outp[g + m] += sg * (aux[g + m] + sv * dv);
      }
    }
  }
}

// ---------------- host-side launcher ----------------
extern "C" void kernel_launch(void* const* d_in, const int* in_sizes, int n_in,
                              void* d_out, int out_size, void* d_ws, size_t ws_size,
                              hipStream_t stream) {
  const float* x       = (const float*)d_in[0];
  const float* w_gate  = (const float*)d_in[1];
  // p1 (half1)
  const float* p1_lnw  = (const float*)d_in[2];
  const float* p1_lnb  = (const float*)d_in[3];
  const float* p1_w1   = (const float*)d_in[4];
  const float* p1_b1   = (const float*)d_in[5];
  const float* p1_w2   = (const float*)d_in[6];
  const float* p1_b2   = (const float*)d_in[7];
  const float* p1_wsca = (const float*)d_in[8];
  const float* p1_bsca = (const float*)d_in[9];
  const float* p1_w3   = (const float*)d_in[10];
  const float* p1_b3   = (const float*)d_in[11];
  // p2 (half2)
  const float* p2_lnw  = (const float*)d_in[12];
  const float* p2_lnb  = (const float*)d_in[13];
  const float* p2_w4   = (const float*)d_in[14];
  const float* p2_b4   = (const float*)d_in[15];
  const float* p2_w5   = (const float*)d_in[16];
  const float* p2_b5   = (const float*)d_in[17];
  // p3 (full): h1, h2, beta, gamma
  const float* h1_lnw  = (const float*)d_in[18];
  const float* h1_lnb  = (const float*)d_in[19];
  const float* h1_w1   = (const float*)d_in[20];
  const float* h1_b1   = (const float*)d_in[21];
  const float* h1_w2   = (const float*)d_in[22];
  const float* h1_b2   = (const float*)d_in[23];
  const float* h1_wsca = (const float*)d_in[24];
  const float* h1_bsca = (const float*)d_in[25];
  const float* h1_w3   = (const float*)d_in[26];
  const float* h1_b3   = (const float*)d_in[27];
  const float* h2_lnw  = (const float*)d_in[28];
  const float* h2_lnb  = (const float*)d_in[29];
  const float* h2_w4   = (const float*)d_in[30];
  const float* h2_b4   = (const float*)d_in[31];
  const float* h2_w5   = (const float*)d_in[32];
  const float* h2_b5   = (const float*)d_in[33];
  const float* beta    = (const float*)d_in[34];
  const float* gamma   = (const float*)d_in[35];

  float* out = (float*)d_out;

  // workspace carve
  char* ws = (char*)d_ws;
  bf16* t_buf = (bf16*)ws;                                  // 32*512*4096 bf16 = 128MB
  bf16* h_buf = (bf16*)(ws + (size_t)134217728);            // 32*256*4096 bf16 = 64MB
  float* y_buf = (float*)(ws + (size_t)134217728 + 67108864);   // f32, 128MB
  float* smallf = (float*)(ws + (size_t)134217728 + 67108864 + 134217728);
  float* gx    = smallf;          // 8192
  float* gates = gx + 8192;       // 96 (pad 128)
  float* pool  = gates + 128;     // 8192
  float* att   = pool + 8192;     // 8192

  const dim3 blk(256);
  const dim3 gGemmA(2048, 4);   // O = 512
  const dim3 gGemmB(2048, 2);   // O = 256

  // out = 0
  k_zero<<<dim3(4096), blk, 0, stream>>>(out, (size_t)out_size);

  // gating
  k_gx<<<dim3(NB * 256), blk, 0, stream>>>(x, gx);
  k_gates<<<dim3(NB), blk, 0, stream>>>(gx, w_gate, gates);

  // ---- expert 0: naf_branch1(x, p1) ----
  k_ln_gemm<<<gGemmA, blk, LNG_SMEM, stream>>>(x, p1_lnw, p1_lnb, p1_w1, p1_b1, t_buf, DWCH);
  k_zero<<<dim3(32), blk, 0, stream>>>(pool, (size_t)(NB * 256));
  k_dwgate<<<dim3(NB * 256, 16), blk, 0, stream>>>(t_buf, p1_w2, p1_b2, h_buf, pool);
  k_sca<<<dim3(NB), blk, 0, stream>>>(pool, p1_wsca, p1_bsca, att);
  k_tail_gemm<<<gGemmB, blk, TAIL_SMEM, stream>>>(h_buf, att, p1_w3, p1_b3, gates,
                                                  nullptr, nullptr, out, 0, 0, 0);

  // ---- expert 1: naf_branch2(x, p2) ----
  k_ln_gemm<<<gGemmA, blk, LNG_SMEM, stream>>>(x, p2_lnw, p2_lnb, p2_w4, p2_b4, t_buf, DWCH);
  k_tail_gemm<<<gGemmB, blk, TAIL_SMEM, stream>>>(t_buf, nullptr, p2_w5, p2_b5, gates,
                                                  nullptr, nullptr, out, 1, 0, 1);

  // ---- expert 2: naf_full(x, p3) ----
  // y = x + beta * branch1(x; h1)
  k_ln_gemm<<<gGemmA, blk, LNG_SMEM, stream>>>(x, h1_lnw, h1_lnb, h1_w1, h1_b1, t_buf, DWCH);
  k_zero<<<dim3(32), blk, 0, stream>>>(pool, (size_t)(NB * 256));
  k_dwgate<<<dim3(NB * 256, 16), blk, 0, stream>>>(t_buf, h1_w2, h1_b2, h_buf, pool);
  k_sca<<<dim3(NB), blk, 0, stream>>>(pool, h1_wsca, h1_bsca, att);
  k_tail_gemm<<<gGemmB, blk, TAIL_SMEM, stream>>>(h_buf, att, h1_w3, h1_b3, gates,
                                                  x, beta, y_buf, 0, 1, 2);
  // out += g2 * (y + gamma * branch2(y; h2))
  k_ln_gemm<<<gGemmA, blk, LNG_SMEM, stream>>>(y_buf, h2_lnw, h2_lnb, h2_w4, h2_b4, t_buf, DWCH);
  k_tail_gemm<<<gGemmB, blk, TAIL_SMEM, stream>>>(t_buf, nullptr, h2_w5, h2_b5, gates,
                                                  y_buf, gamma, out, 1, 2, 2);
}
